// ScaledDotProductAttention_22840636080983
// MI455X (gfx1250) — matrix operand order repaired and checked by reading, not compiled, not run on hardware
//
#include <hip/hip_runtime.h>
#include <math.h>

typedef __attribute__((ext_vector_type(16))) __bf16 bf16x16;
typedef __attribute__((ext_vector_type(8)))  float  f32x8;
typedef __attribute__((ext_vector_type(4)))  int    i32x4;
typedef __attribute__((ext_vector_type(2)))  int    i32x2;

#define B_DIM 2
#define H_DIM 16
#define S_DIM 2048
#define D_DIM 64
#define SCALE 0.125f   // 1/sqrt(64)

#define KL_PAD 72      // K-tile LDS row stride in bf16 (36 dwords -> conflict-free)
#define VT_PAD 40      // V^T-tile LDS row stride in bf16 (20 dwords -> conflict-free)

// Build 16x32 bf16 A-fragment from an LDS row: two contiguous 16B reads.
// element e -> col = 2*hb + e (contiguous 16 per lane half)
__device__ __forceinline__ bf16x16 lds_frag(const __bf16* rowp, int hb) {
  union { bf16x16 v; i32x4 q[2]; } u;
  u.q[0] = *(const i32x4*)(rowp + 2 * hb);
  u.q[1] = *(const i32x4*)(rowp + 2 * hb + 8);
  return u.v;
}

__global__ __launch_bounds__(128, 1)
void fa_fwd(const float* __restrict__ Q, const float* __restrict__ V,
            const float* __restrict__ K, const float* __restrict__ AM,
            const float* __restrict__ SM, float* __restrict__ O) {
  // Double-buffered staging tiles (~19.5 KB of 320 KB WGP LDS)
  __shared__ __align__(16) __bf16 Kl[2][32][KL_PAD];  // [buf][key n][d]
  __shared__ __align__(16) __bf16 Vt[2][64][VT_PAD];  // [buf][d][key n]

  const int tid  = threadIdx.x;
  const int lane = tid & 31;
  const int wave = tid >> 5;
  const int qb   = blockIdx.x & 15;   // 16 q-blocks of 128 rows
  const int bh   = blockIdx.x >> 4;   // 0..31 (b*16 + h)
  const int b    = bh >> 4;
  const int q0   = qb * 128 + wave * 32;   // 32 query rows per wave (2 m-tiles)
  const int c    = lane & 15;
  const int half = lane >> 4;
  const int hb   = half << 3;         // 0 or 8

  const float* Qh   = Q  + (size_t)bh * (S_DIM * D_DIM);
  const float* Kh   = K  + (size_t)bh * (S_DIM * D_DIM);
  const float* Vh   = V  + (size_t)bh * (S_DIM * D_DIM);
  const float* Srow = SM + (size_t)b * S_DIM;

  const float* Mrow[2];
  float*       Orow[2];
  bf16x16      bq[2][2];   // [m-tile][d-chunk] B operand: Q^T, lane = column m
#pragma unroll
  for (int mt = 0; mt < 2; ++mt) {
    const int qr = q0 + mt * 16 + c;
    Mrow[mt] = AM + (size_t)b * (size_t)S_DIM * S_DIM + (size_t)qr * S_DIM;
    Orow[mt] = O  + (size_t)bh * (S_DIM * D_DIM) + (size_t)qr * D_DIM;
    const float* qrow = Qh + (size_t)qr * D_DIM + 16 * half;
#pragma unroll
    for (int e = 0; e < 16; ++e) {
      bq[mt][0][e] = (__bf16)qrow[e];        // d 0..31
      bq[mt][1][e] = (__bf16)qrow[32 + e];   // d 32..63
    }
  }

  // Staging assignments (whole WG, 128 threads)
  const int kn  = tid & 31;           // K: key row
  const int kds = (tid >> 5) << 4;    // K: d-segment (0,16,32,48)
  const int vj  = tid & 7;            // V: key quad (rows 4vj..4vj+3)
  const int vdb = (tid >> 3) << 2;    // V: d base (0..60 step 4)

  // Stage one 32-key block (kb) into buffer p: K as bf16 [n][d],
  // V as bf16 transposed [d][n] (transpose done in packed ds_store_b64s).
  auto stage = [&](int p, int kb) {
    {
      const float4* src = (const float4*)(Kh + (size_t)(kb + kn) * D_DIM + kds);
      float4 f0 = src[0], f1 = src[1], f2 = src[2], f3 = src[3];
      union { __bf16 h[16]; i32x4 q[2]; } pk;
      pk.h[0]=(__bf16)f0.x;  pk.h[1]=(__bf16)f0.y;  pk.h[2]=(__bf16)f0.z;  pk.h[3]=(__bf16)f0.w;
      pk.h[4]=(__bf16)f1.x;  pk.h[5]=(__bf16)f1.y;  pk.h[6]=(__bf16)f1.z;  pk.h[7]=(__bf16)f1.w;
      pk.h[8]=(__bf16)f2.x;  pk.h[9]=(__bf16)f2.y;  pk.h[10]=(__bf16)f2.z; pk.h[11]=(__bf16)f2.w;
      pk.h[12]=(__bf16)f3.x; pk.h[13]=(__bf16)f3.y; pk.h[14]=(__bf16)f3.z; pk.h[15]=(__bf16)f3.w;
      *(i32x4*)&Kl[p][kn][kds]     = pk.q[0];
      *(i32x4*)&Kl[p][kn][kds + 8] = pk.q[1];
    }
    {
      const float* v0 = Vh + (size_t)(kb + 4 * vj) * D_DIM + vdb;
      float4 r0 = *(const float4*)(v0);
      float4 r1 = *(const float4*)(v0 + D_DIM);
      float4 r2 = *(const float4*)(v0 + 2 * D_DIM);
      float4 r3 = *(const float4*)(v0 + 3 * D_DIM);
      float a0[4] = {r0.x, r0.y, r0.z, r0.w};
      float a1[4] = {r1.x, r1.y, r1.z, r1.w};
      float a2[4] = {r2.x, r2.y, r2.z, r2.w};
      float a3[4] = {r3.x, r3.y, r3.z, r3.w};
#pragma unroll
      for (int d = 0; d < 4; ++d) {
        union { __bf16 h[4]; i32x2 q; } w;
        w.h[0] = (__bf16)a0[d];
        w.h[1] = (__bf16)a1[d];
        w.h[2] = (__bf16)a2[d];
        w.h[3] = (__bf16)a3[d];
        *(i32x2*)&Vt[p][vdb + d][4 * vj] = w.q;
      }
    }
  };

  f32x8 zc = {};
  f32x8 acc[2][4];
#pragma unroll
  for (int mt = 0; mt < 2; ++mt)
#pragma unroll
    for (int t = 0; t < 4; ++t) acc[mt][t] = zc;
  float m_run[2] = {-INFINITY, -INFINITY};
  float l_run[2] = {0.0f, 0.0f};

  stage(0, 0);   // prologue: fill buffer 0

  for (int kb = 0; kb < S_DIM; kb += 32) {
    const int p = (kb >> 5) & 1;
    __syncthreads();   // buffer p staged; previous reads of buffer p^1 done

    // ---- overlap: stage next block into buffer p^1 while computing on p ----
    if (kb + 32 < S_DIM) {
      stage(p ^ 1, kb + 32);
      if (kb + 64 < S_DIM) {
        const float* kn2 = Kh + (size_t)(kb + 64 + lane) * D_DIM;
        const float* vn2 = Vh + (size_t)(kb + 64 + lane) * D_DIM;
        __builtin_prefetch(kn2, 0, 1);
        __builtin_prefetch(kn2 + 32, 0, 1);
        __builtin_prefetch(vn2, 0, 1);
        __builtin_prefetch(vn2 + 32, 0, 1);
      }
    }

    // ---- seq_mask segment (shared by both m-tiles) ----
    float sq_lo[8], sq_hi[8];
#pragma unroll
    for (int i = 0; i < 8; ++i) {
      sq_lo[i] = Srow[kb + hb + i];
      sq_hi[i] = Srow[kb + 16 + hb + i];
    }

    // ---- K A-fragments from LDS (shared by both m-tiles) ----
    bf16x16 akl0 = lds_frag(&Kl[p][c][0],       hb);
    bf16x16 akl1 = lds_frag(&Kl[p][c][32],      hb);
    bf16x16 akh0 = lds_frag(&Kl[p][16 + c][0],  hb);
    bf16x16 akh1 = lds_frag(&Kl[p][16 + c][32], hb);

    bf16x16 bp[2];   // P^T B operands, one per m-tile
#pragma unroll
    for (int mt = 0; mt < 2; ++mt) {
      // ---- scores: S^T = K_tile (A) x Q^T (B) ----
      f32x8 slo = __builtin_amdgcn_wmma_f32_16x16x32_bf16(false, akl0, false, bq[mt][0], (short)0, zc,  false, false);
      slo       = __builtin_amdgcn_wmma_f32_16x16x32_bf16(false, akl1, false, bq[mt][1], (short)0, slo, false, false);
      f32x8 shi = __builtin_amdgcn_wmma_f32_16x16x32_bf16(false, akh0, false, bq[mt][0], (short)0, zc,  false, false);
      shi       = __builtin_amdgcn_wmma_f32_16x16x32_bf16(false, akh1, false, bq[mt][1], (short)0, shi, false, false);

      float sl[8], sh[8];
#pragma unroll
      for (int i = 0; i < 8; ++i) {
        sl[i] = slo[i] * SCALE + Mrow[mt][kb + hb + i];
        sh[i] = shi[i] * SCALE + Mrow[mt][kb + 16 + hb + i];
      }

      // ---- online softmax (per-lane scalar state: lane <-> query column m) ----
      float bm = sl[0];
#pragma unroll
      for (int i = 0; i < 8; ++i) { bm = fmaxf(bm, sl[i]); bm = fmaxf(bm, sh[i]); }
      bm = fmaxf(bm, __shfl_xor(bm, 16, 32));
      const float mnew = fmaxf(m_run[mt], bm);

      float pl[8], ph[8], ps = 0.0f;
#pragma unroll
      for (int i = 0; i < 8; ++i) {
        pl[i] = __expf(sl[i] - mnew);
        ph[i] = __expf(sh[i] - mnew);
        ps += pl[i] + ph[i];
      }
      ps += __shfl_xor(ps, 16, 32);

      const float alpha = __expf(m_run[mt] - mnew);
      l_run[mt] = l_run[mt] * alpha + ps;   // denominator WITHOUT seq_mask
      m_run[mt] = mnew;
#pragma unroll
      for (int t = 0; t < 4; ++t)
#pragma unroll
        for (int i = 0; i < 8; ++i) acc[mt][t][i] *= alpha;

      // ---- post-softmax seq_mask + pack P^T as B operand ----
      union { bf16x16 v; int u[8]; } own, bpk;
#pragma unroll
      for (int i = 0; i < 8; ++i) {
        own.v[i]     = (__bf16)(pl[i] * sq_lo[i]);
        own.v[8 + i] = (__bf16)(ph[i] * sq_hi[i]);
      }
      int part[8];
#pragma unroll
      for (int pp = 0; pp < 8; ++pp) part[pp] = __shfl_xor(own.u[pp], 16, 32);
#pragma unroll
      for (int pp = 0; pp < 4; ++pp) {
        bpk.u[pp]     = half ? part[4 + pp] : own.u[pp];
        bpk.u[4 + pp] = half ? own.u[4 + pp] : part[pp];
      }
      bp[mt] = bpk.v;
    }

    // ---- O^T += V^T (A) x P^T (B): V fragments from LDS, 2 ds_load_b128 each ----
#pragma unroll
    for (int t = 0; t < 4; ++t) {
      bf16x16 av = lds_frag(&Vt[p][16 * t + c][0], hb);
      acc[0][t] = __builtin_amdgcn_wmma_f32_16x16x32_bf16(false, av, false, bp[0], (short)0, acc[0][t], false, false);
      acc[1][t] = __builtin_amdgcn_wmma_f32_16x16x32_bf16(false, av, false, bp[1], (short)0, acc[1][t], false, false);
    }
  }

  // ---- normalize and store: O^T tile rows d = 16t + hb + i, column m = c ----
#pragma unroll
  for (int mt = 0; mt < 2; ++mt) {
    const float inv = 1.0f / l_run[mt];
#pragma unroll
    for (int t = 0; t < 4; ++t)
#pragma unroll
      for (int i = 0; i < 8; ++i)
        Orow[mt][t * 16 + hb + i] = acc[mt][t][i] * inv;
  }
}

extern "C" void kernel_launch(void* const* d_in, const int* in_sizes, int n_in,
                              void* d_out, int out_size, void* d_ws, size_t ws_size,
                              hipStream_t stream) {
  // setup_inputs order: q, v, k, att_mask, seq_mask
  const float* q  = (const float*)d_in[0];
  const float* v  = (const float*)d_in[1];
  const float* k  = (const float*)d_in[2];
  const float* am = (const float*)d_in[3];
  const float* sm = (const float*)d_in[4];
  float* out = (float*)d_out;

  dim3 grid(B_DIM * H_DIM * (S_DIM / 128));  // 512 workgroups
  dim3 block(128);                           // 4 wave32s, 32 q-rows each
  fa_fwd<<<grid, block, 0, stream>>>(q, v, k, am, sm, out);

  (void)in_sizes; (void)n_in; (void)out_size; (void)d_ws; (void)ws_size;
}
